// MoeRowParallelLinear_12000138625734
// MI455X (gfx1250) — matrix operand order repaired and checked by reading, not compiled, not run on hardware
//
#include <hip/hip_runtime.h>

#define NEXP   8
#define IN_F   2048
#define OUT_F  2048
#define T_TOK  16384
#define BM     128
#define BN     128
#define BK     32

typedef __attribute__((ext_vector_type(16))) __bf16 v16bf;
typedef __attribute__((ext_vector_type(8)))  __bf16 v8bf;
typedef __attribute__((ext_vector_type(4)))  __bf16 v4bf;
typedef __attribute__((ext_vector_type(8)))  float  v8f;
typedef __attribute__((ext_vector_type(4)))  float  v4f;
typedef __attribute__((ext_vector_type(4)))  int    v4i;

union Frag16 { v16bf v; v8bf h[2]; };

#if __has_builtin(__builtin_amdgcn_global_load_async_to_lds_b128)
#define HAVE_ASYNC_LDS 1
#endif

// builtin signature: (v4i addrspace(1)*, v4i addrspace(3)*, imm offset, imm cpol)
typedef __attribute__((address_space(1))) v4i glob_v4i;
typedef __attribute__((address_space(3))) v4i lds_v4i;

__device__ __forceinline__ void async_copy16(const __bf16* gsrc, __bf16* ldst) {
#ifdef HAVE_ASYNC_LDS
    __builtin_amdgcn_global_load_async_to_lds_b128(
        (glob_v4i*)gsrc, (lds_v4i*)ldst, 0, 0);
#else
    *(v8bf*)ldst = *(const v8bf*)gsrc;      // VGPR round-trip fallback
#endif
}

__device__ __forceinline__ void wait_async_copies() {
#ifdef HAVE_ASYNC_LDS
#if __has_builtin(__builtin_amdgcn_s_wait_asynccnt)
    __builtin_amdgcn_s_wait_asynccnt(0);
#else
    asm volatile("s_wait_asynccnt 0" ::: "memory");
#endif
#endif
}

__device__ __forceinline__ int find_expert(const long long* __restrict__ off, int tok) {
    int e = 0;
#pragma unroll
    for (int i = 1; i < NEXP; ++i) e += (tok >= (int)off[i]);
    return e;
}

// ---------------------------------------------------------------------------
// Pre-pass: split fp32 -> bf16 (hi) + bf16 (residual) planes, k-contiguous.
// Pure bandwidth: ~536 MB total traffic ≈ 23 us at 23.3 TB/s.
// ---------------------------------------------------------------------------
__global__ __launch_bounds__(256)
void split_bf16_planes(const float* __restrict__ src,
                       __bf16* __restrict__ hi,
                       __bf16* __restrict__ lo,
                       long long n4)              // number of float4 chunks
{
    long long i = (long long)blockIdx.x * blockDim.x + threadIdx.x;
    const long long stride = (long long)gridDim.x * blockDim.x;
    for (; i < n4; i += stride) {
        v4f f = ((const v4f*)src)[i];
        v4bf h, l;
#pragma unroll
        for (int q = 0; q < 4; ++q) {
            float x = f[q];
            __bf16 hq = (__bf16)x;
            h[q] = hq;
            l[q] = (__bf16)(x - (float)hq);
        }
        ((v4bf*)hi)[i] = h;
        ((v4bf*)lo)[i] = l;
    }
}

// ---------------------------------------------------------------------------
// Main grouped GEMM: bf16x3 WMMA, double-buffered async global->LDS staging.
// ---------------------------------------------------------------------------
__global__ __launch_bounds__(256)
void moe_main_async(const __bf16* __restrict__ Xh, const __bf16* __restrict__ Xl,
                    const __bf16* __restrict__ Wh, const __bf16* __restrict__ Wl,
                    const long long* __restrict__ eoff,
                    const float* __restrict__ Bias,
                    float* __restrict__ Y)
{
    // [buf][plane][row][k]  plane: 0=A_hi 1=A_lo 2=B_hi 3=B_lo   (64 KB)
    __shared__ __bf16 smem[2][4][BM][BK];
    __shared__ int s_eid[BM];

    const int tid  = threadIdx.x;
    const int lane = tid & 31;
    const int w    = tid >> 5;
    const int wm   = w & 1;          // 2 waves along M
    const int wn   = w >> 1;         // 4 waves along N
    const int n0   = blockIdx.x * BN;
    const int t0   = blockIdx.y * BM;

    if (tid < BM) s_eid[tid] = find_expert(eoff, t0 + tid);
    __syncthreads();

    const int e_lo   = s_eid[0];
    const int e_hi   = s_eid[BM - 1];
    const bool mixed = (e_lo != e_hi);   // wave-uniform: boundary tile?

    v8f acc[4][2];
#pragma unroll
    for (int s = 0; s < 4; ++s)
#pragma unroll
        for (int t = 0; t < 2; ++t) acc[s][t] = (v8f){0.f,0.f,0.f,0.f,0.f,0.f,0.f,0.f};

    // WMMA bf16 fragment addressing (CDNA5 VGPR layouts)
    const int kbA  = (lane < 16) ? 0 : 8;
    const int kbB  = (lane < 16) ? 0 : 8;
    const int lrow = lane & 15;

    // staging: thread -> two 16B chunks per plane (512 chunks of 16B per 8KB plane)
    const int cid0 = tid * 2;

    for (int e = e_lo; e <= e_hi; ++e) {
        const size_t woff = (size_t)e * (size_t)OUT_F * (size_t)IN_F;
        const __bf16* WhE = Wh + woff;
        const __bf16* WlE = Wl + woff;

        bool am[4];
#pragma unroll
        for (int s = 0; s < 4; ++s)
            am[s] = (s_eid[wm * 64 + s * 16 + lrow] == e);

        auto stage = [&](int buf, int kb) {
#pragma unroll
            for (int c = 0; c < 2; ++c) {
                const int cid = cid0 + c;
                const int row = cid >> 2;
                const int ko  = (cid & 3) * 8;
                const size_t ga = (size_t)(t0 + row) * IN_F + kb + ko;
                const size_t gb = (size_t)(n0 + row) * IN_F + kb + ko;
                async_copy16(Xh  + ga, &smem[buf][0][row][ko]);
                async_copy16(Xl  + ga, &smem[buf][1][row][ko]);
                async_copy16(WhE + gb, &smem[buf][2][row][ko]);
                async_copy16(WlE + gb, &smem[buf][3][row][ko]);
            }
        };

        __syncthreads();           // previous reads of smem complete before refill
        stage(0, 0);
        int cur = 0;

        for (int kb = 0; kb < IN_F; kb += BK) {
            wait_async_copies();   // our copies for `cur` landed
            __syncthreads();       // everyone's copies landed; prior compute done
            if (kb + BK < IN_F) stage(cur ^ 1, kb + BK);

            // ---- B fragments ----
            Frag16 bh[2], bl[2];
#pragma unroll
            for (int t = 0; t < 2; ++t) {
                const int n = wn * 32 + t * 16 + lrow;
                bh[t].h[0] = *(const v8bf*)&smem[cur][2][n][kbB];
                bh[t].h[1] = *(const v8bf*)&smem[cur][2][n][kbB + 16];
                bl[t].h[0] = *(const v8bf*)&smem[cur][3][n][kbB];
                bl[t].h[1] = *(const v8bf*)&smem[cur][3][n][kbB + 16];
            }

            // ---- A fragments + bf16x3 accumulation ----
#pragma unroll
            for (int s = 0; s < 4; ++s) {
                const int r = wm * 64 + s * 16 + lrow;
                Frag16 ah, al;
                ah.h[0] = *(const v8bf*)&smem[cur][0][r][kbA];
                ah.h[1] = *(const v8bf*)&smem[cur][0][r][kbA + 16];
                al.h[0] = *(const v8bf*)&smem[cur][1][r][kbA];
                al.h[1] = *(const v8bf*)&smem[cur][1][r][kbA + 16];
                if (mixed) {                       // uniform branch: fast path skips
                    if (!am[s]) { v16bf z = {}; ah.v = z; al.v = z; }
                }
#pragma unroll
                for (int t = 0; t < 2; ++t) {
                    acc[s][t] = __builtin_amdgcn_wmma_f32_16x16x32_bf16(
                        false, ah.v, false, bh[t].v, (short)0, acc[s][t], false, false);
                    acc[s][t] = __builtin_amdgcn_wmma_f32_16x16x32_bf16(
                        false, ah.v, false, bl[t].v, (short)0, acc[s][t], false, false);
                    acc[s][t] = __builtin_amdgcn_wmma_f32_16x16x32_bf16(
                        false, al.v, false, bh[t].v, (short)0, acc[s][t], false, false);
                }
            }
            cur ^= 1;
        }
    }

    // ---- epilogue: add per-token expert bias, store ----
    const int colBase = n0 + wn * 32 + lrow;
    const int rOff    = (lane >> 4) * 8;
#pragma unroll
    for (int s = 0; s < 4; ++s) {
#pragma unroll
        for (int t = 0; t < 2; ++t) {
            const int col = colBase + t * 16;
#pragma unroll
            for (int i = 0; i < 8; ++i) {
                const int lr  = wm * 64 + s * 16 + rOff + i;
                const int row = t0 + lr;
                const int eid = s_eid[lr];
                Y[(size_t)row * OUT_F + col] = acc[s][t][i] + Bias[eid * OUT_F + col];
            }
        }
    }
}

// ---------------------------------------------------------------------------
// Fallback (no workspace): converts in-loop.
// ---------------------------------------------------------------------------
__global__ __launch_bounds__(256)
void moe_fallback(const float* __restrict__ X,
                  const long long* __restrict__ eoff,
                  const float* __restrict__ W,
                  const float* __restrict__ Bias,
                  float* __restrict__ Y)
{
    __shared__ __bf16 Ah[BM][BK];
    __shared__ __bf16 Al[BM][BK];
    __shared__ __bf16 Bh[BN][BK];
    __shared__ __bf16 Bl[BN][BK];
    __shared__ int s_eid[BM];

    const int tid  = threadIdx.x;
    const int lane = tid & 31;
    const int w    = tid >> 5;
    const int wm   = w & 1;
    const int wn   = w >> 1;
    const int n0   = blockIdx.x * BN;
    const int t0   = blockIdx.y * BM;

    if (tid < BM) s_eid[tid] = find_expert(eoff, t0 + tid);
    __syncthreads();

    const int e_lo   = s_eid[0];
    const int e_hi   = s_eid[BM - 1];
    const bool mixed = (e_lo != e_hi);

    v8f acc[4][2];
#pragma unroll
    for (int s = 0; s < 4; ++s)
#pragma unroll
        for (int t = 0; t < 2; ++t) acc[s][t] = (v8f){0.f,0.f,0.f,0.f,0.f,0.f,0.f,0.f};

    const int srow = tid >> 1;
    const int skc  = (tid & 1) * 16;
    const int kbA  = (lane < 16) ? 0 : 8;
    const int kbB  = (lane < 16) ? 0 : 8;
    const int lrow = lane & 15;

    for (int e = e_lo; e <= e_hi; ++e) {
        const float* We = W + (size_t)e * (size_t)OUT_F * (size_t)IN_F;
        bool am[4];
#pragma unroll
        for (int s = 0; s < 4; ++s)
            am[s] = (s_eid[wm * 64 + s * 16 + lrow] == e);

        const float* srcA = X  + (size_t)(t0 + srow) * IN_F + skc;
        const float* srcB = We + (size_t)(n0 + srow) * IN_F + skc;

        for (int kb = 0; kb < IN_F; kb += BK) {
            __syncthreads();
#pragma unroll
            for (int half = 0; half < 2; ++half) {
                v4f f0 = *(const v4f*)(srcA + kb + half * 8);
                v4f f1 = *(const v4f*)(srcA + kb + half * 8 + 4);
                v8bf hh, ll;
#pragma unroll
                for (int q = 0; q < 4; ++q) {
                    float x = f0[q]; __bf16 h = (__bf16)x;
                    hh[q] = h;      ll[q] = (__bf16)(x - (float)h);
                    float y = f1[q]; __bf16 h2 = (__bf16)y;
                    hh[4 + q] = h2; ll[4 + q] = (__bf16)(y - (float)h2);
                }
                *(v8bf*)&Ah[srow][skc + half * 8] = hh;
                *(v8bf*)&Al[srow][skc + half * 8] = ll;
            }
#pragma unroll
            for (int half = 0; half < 2; ++half) {
                v4f f0 = *(const v4f*)(srcB + kb + half * 8);
                v4f f1 = *(const v4f*)(srcB + kb + half * 8 + 4);
                v8bf hh, ll;
#pragma unroll
                for (int q = 0; q < 4; ++q) {
                    float x = f0[q]; __bf16 h = (__bf16)x;
                    hh[q] = h;      ll[q] = (__bf16)(x - (float)h);
                    float y = f1[q]; __bf16 h2 = (__bf16)y;
                    hh[4 + q] = h2; ll[4 + q] = (__bf16)(y - (float)h2);
                }
                *(v8bf*)&Bh[srow][skc + half * 8] = hh;
                *(v8bf*)&Bl[srow][skc + half * 8] = ll;
            }
            __syncthreads();

            Frag16 bh[2], bl[2];
#pragma unroll
            for (int t = 0; t < 2; ++t) {
                const int n = wn * 32 + t * 16 + lrow;
                bh[t].h[0] = *(const v8bf*)&Bh[n][kbB];
                bh[t].h[1] = *(const v8bf*)&Bh[n][kbB + 16];
                bl[t].h[0] = *(const v8bf*)&Bl[n][kbB];
                bl[t].h[1] = *(const v8bf*)&Bl[n][kbB + 16];
            }
#pragma unroll
            for (int s = 0; s < 4; ++s) {
                const int r = wm * 64 + s * 16 + lrow;
                Frag16 ah, al;
                ah.h[0] = *(const v8bf*)&Ah[r][kbA];
                ah.h[1] = *(const v8bf*)&Ah[r][kbA + 16];
                al.h[0] = *(const v8bf*)&Al[r][kbA];
                al.h[1] = *(const v8bf*)&Al[r][kbA + 16];
                if (mixed) {
                    if (!am[s]) { v16bf z = {}; ah.v = z; al.v = z; }
                }
#pragma unroll
                for (int t = 0; t < 2; ++t) {
                    acc[s][t] = __builtin_amdgcn_wmma_f32_16x16x32_bf16(
                        false, ah.v, false, bh[t].v, (short)0, acc[s][t], false, false);
                    acc[s][t] = __builtin_amdgcn_wmma_f32_16x16x32_bf16(
                        false, ah.v, false, bl[t].v, (short)0, acc[s][t], false, false);
                    acc[s][t] = __builtin_amdgcn_wmma_f32_16x16x32_bf16(
                        false, al.v, false, bh[t].v, (short)0, acc[s][t], false, false);
                }
            }
        }
    }

    const int colBase = n0 + wn * 32 + lrow;
    const int rOff    = (lane >> 4) * 8;
#pragma unroll
    for (int s = 0; s < 4; ++s) {
#pragma unroll
        for (int t = 0; t < 2; ++t) {
            const int col = colBase + t * 16;
#pragma unroll
            for (int i = 0; i < 8; ++i) {
                const int lr  = wm * 64 + s * 16 + rOff + i;
                const int row = t0 + lr;
                const int eid = s_eid[lr];
                Y[(size_t)row * OUT_F + col] = acc[s][t][i] + Bias[eid * OUT_F + col];
            }
        }
    }
}

extern "C" void kernel_launch(void* const* d_in, const int* in_sizes, int n_in,
                              void* d_out, int out_size, void* d_ws, size_t ws_size,
                              hipStream_t stream) {
    const float*     X    = (const float*)d_in[0];
    const long long* eoff = (const long long*)d_in[1];
    const float*     W    = (const float*)d_in[2];
    const float*     Bias = (const float*)d_in[3];
    float*           Y    = (float*)d_out;

    const size_t nX   = (size_t)T_TOK * IN_F;                 // 33.55M
    const size_t nW   = (size_t)NEXP * OUT_F * IN_F;          // 33.55M
    const size_t need = 2 * (nX + nW) * 2;                    // 268.4 MB

    dim3 grid(OUT_F / BN, T_TOK / BM);

    if (ws_size >= need) {
        char* ws = (char*)d_ws;
        __bf16* Xh = (__bf16*)(ws);
        __bf16* Xl = (__bf16*)(ws + nX * 2);
        __bf16* Wh = (__bf16*)(ws + nX * 4);
        __bf16* Wl = (__bf16*)(ws + nX * 4 + nW * 2);

        split_bf16_planes<<<8192, 256, 0, stream>>>(X, Xh, Xl, (long long)(nX / 4));
        split_bf16_planes<<<8192, 256, 0, stream>>>(W, Wh, Wl, (long long)(nW / 4));
        moe_main_async<<<grid, 256, 0, stream>>>(Xh, Xl, Wh, Wl, eoff, Bias, Y);
    } else {
        moe_fallback<<<grid, 256, 0, stream>>>(X, eoff, W, Bias, Y);
    }
}
